// GCNRegressor_57801669870214
// MI455X (gfx1250) — matrix operand order repaired and checked by reading, not compiled, not run on hardware
//
#include <hip/hip_runtime.h>

// ---------------------------------------------------------------------------
// GCN 3-layer + mean-pool + linear head for MI455X (gfx1250).
// Dense transforms use V_WMMA_F32_16X16X32_F16 (wave32 WMMA); edge
// aggregation uses coalesced global f32 atomics (L2-resident: 2x51MB
// activation buffers fit in the 192MB L2). Self-loops are folded into the
// per-node init pass so no atomics are spent on them.
// ---------------------------------------------------------------------------

typedef __attribute__((ext_vector_type(16))) _Float16 v16h;
typedef __attribute__((ext_vector_type(8)))  float    v8f;
typedef __attribute__((ext_vector_type(2)))  _Float16 h2;

#define HDIM 128
#define LROW 136   // LDS row stride in halfs (pad: lane m hits bank 4m)

// ---- W[k][n] (f32, row-major) -> Wt[n][k] (f16) so B-fragment K-runs are
// contiguous 16-byte loads (merge into global_load_b128).
__global__ __launch_bounds__(128) void wt_convert_kernel(
    const float* __restrict__ W, _Float16* __restrict__ Wt) {
  int k = blockIdx.x;      // 0..127
  int n = threadIdx.x;     // 0..127
  Wt[n * HDIM + k] = (_Float16)W[k * HDIM + n];
}

// ---- deg[n] = 1.0 (self-loop weight folded in here)
__global__ __launch_bounds__(256) void deg_init_kernel(float* __restrict__ deg, int N) {
  int i = blockIdx.x * blockDim.x + threadIdx.x;
  if (i < N) deg[i] = 1.0f;
}

// ---- deg[col[e]] += ew[e]
__global__ __launch_bounds__(256) void deg_accum_kernel(
    const long long* __restrict__ col, const float* __restrict__ ew,
    float* __restrict__ deg, int E) {
  int e = blockIdx.x * blockDim.x + threadIdx.x;
  if (e < E) atomicAdd(&deg[(int)col[e]], ew[e]);
}

// ---- dinv = rsqrt(deg) (deg >= 1 always); selfnorm = dinv^2
__global__ __launch_bounds__(256) void dinv_kernel(
    const float* __restrict__ deg, float* __restrict__ dinv,
    float* __restrict__ selfn, int N) {
  int i = blockIdx.x * blockDim.x + threadIdx.x;
  if (i < N) {
    float dv = rsqrtf(deg[i]);
    dinv[i] = dv;
    selfn[i] = dv * dv;
  }
}

// ---- norm[e] = dinv[row]*ew*dinv[col]
__global__ __launch_bounds__(256) void norm_kernel(
    const long long* __restrict__ row, const long long* __restrict__ col,
    const float* __restrict__ ew, const float* __restrict__ dinv,
    float* __restrict__ norm, int E) {
  int e = blockIdx.x * blockDim.x + threadIdx.x;
  if (e < E) norm[e] = dinv[(int)row[e]] * ew[e] * dinv[(int)col[e]];
}

// ---------------------------------------------------------------------------
// t = (relu_in ? relu(h) : h) @ W     via WMMA f16 -> f32.
// Block = 256 threads = 8 waves; each block owns a 16-row tile, wave w owns
// output columns [16w, 16w+16). A tile staged through LDS (f32->f16 convert
// fused, ReLU of the previous layer fused), K=128 consumed as 4 WMMA steps.
// ---------------------------------------------------------------------------
__global__ __launch_bounds__(256) void gcn_gemm_wmma_kernel(
    const float* __restrict__ hin, const _Float16* __restrict__ Wt,
    float* __restrict__ outp, int relu_in, int nrows) {
  __shared__ _Float16 Atile[16 * LROW];

  const int tid = threadIdx.x;
  const int rbase = blockIdx.x << 4;
  const bool full_tile = (rbase + 16 <= nrows);   // block-uniform

  // ---- stage A: 16x128 f32 -> f16 in LDS, 8 elements/thread
  {
    int r = tid >> 4;            // 0..15
    int c = (tid & 15) << 3;     // 0,8,...,120
    int grow = rbase + r;
    float4 f0 = make_float4(0.f, 0.f, 0.f, 0.f), f1 = f0;
    if (grow < nrows) {
      const float4* src = (const float4*)(hin + (size_t)grow * HDIM + c);
      f0 = src[0];
      f1 = src[1];
    }
    if (relu_in) {
      f0.x = fmaxf(f0.x, 0.f); f0.y = fmaxf(f0.y, 0.f);
      f0.z = fmaxf(f0.z, 0.f); f0.w = fmaxf(f0.w, 0.f);
      f1.x = fmaxf(f1.x, 0.f); f1.y = fmaxf(f1.y, 0.f);
      f1.z = fmaxf(f1.z, 0.f); f1.w = fmaxf(f1.w, 0.f);
    }
    _Float16* dst = &Atile[r * LROW + c];
    dst[0] = (_Float16)f0.x; dst[1] = (_Float16)f0.y;
    dst[2] = (_Float16)f0.z; dst[3] = (_Float16)f0.w;
    dst[4] = (_Float16)f1.x; dst[5] = (_Float16)f1.y;
    dst[6] = (_Float16)f1.z; dst[7] = (_Float16)f1.w;
  }
  __syncthreads();

  const int lane   = tid & 31;
  const int wv     = tid >> 5;        // 0..7 -> column tile
  const int halfid = lane >> 4;       // 0 | 1
  const int mn     = lane & 15;       // A row / B,D column

  const _Float16* wcol = Wt + ((size_t)(wv * 16 + mn) << 7);  // Wt[n][*]

  v8f acc = {};
#pragma unroll
  for (int kt = 0; kt < 4; ++kt) {
    v16h a, b;
#pragma unroll
    for (int i = 0; i < 8; ++i) {
      // A k order used here (contiguous, matches B): dword i K-pair = 2i + (half?16:0)
      // (one contiguous 32B run per lane -> ds_load_b128 pair)
      const int ka = kt * 32 + 2 * i + (halfid ? 16 : 0);
      // B k order used here (contiguous, matches A): dword i K-pair = 2i + (half?16:0)
      // (one contiguous 32B run per lane -> global_load_b128 pair)
      const int kb = kt * 32 + 2 * i + (halfid ? 16 : 0);
      h2 av = *(const h2*)(&Atile[mn * LROW + ka]);
      h2 bv = *(const h2*)(&wcol[kb]);
      a[2 * i] = av.x; a[2 * i + 1] = av.y;
      b[2 * i] = bv.x; b[2 * i + 1] = bv.y;
    }
    acc = __builtin_amdgcn_wmma_f32_16x16x32_f16(
        /*neg_a=*/false, a, /*neg_b=*/false, b,
        /*c_mod=*/(short)0, acc, /*reuse_a=*/false, /*reuse_b=*/false);
  }

  // D layout: VGPR r -> row (r + 8*half), column mn.
  // Fast path: one base address + 8 immediate-offset stores (no exec churn).
  const int col0 = wv * 16 + mn;
  if (full_tile) {
    float* orow = outp + (size_t)(rbase + 8 * halfid) * HDIM + col0;
#pragma unroll
    for (int r = 0; r < 8; ++r) orow[(size_t)r * HDIM] = acc[r];
  } else {
#pragma unroll
    for (int r = 0; r < 8; ++r) {
      int grow = rbase + r + 8 * halfid;
      if (grow < nrows) outp[(size_t)grow * HDIM + col0] = acc[r];
    }
  }
}

// ---- agg[n,:] = b + selfnorm[n] * t[n,:]   (bias + self-loop message)
// float4-vectorized: pure-bandwidth pass.
__global__ __launch_bounds__(256) void agg_init_kernel(
    const float* __restrict__ t, const float* __restrict__ bias,
    const float* __restrict__ selfn, float* __restrict__ agg, int N) {
  int idx = blockIdx.x * blockDim.x + threadIdx.x;   // over N*32 float4s
  if (idx >= N * (HDIM / 4)) return;
  int n = idx >> 5;                 // node
  int c4 = (idx & 31) << 2;         // channel base
  float s = selfn[n];
  float4 tv = ((const float4*)t)[idx];
  const float4 bv = *(const float4*)(bias + c4);
  float4 o;
  o.x = bv.x + s * tv.x;
  o.y = bv.y + s * tv.y;
  o.z = bv.z + s * tv.z;
  o.w = bv.w + s * tv.w;
  ((float4*)agg)[idx] = o;
}

// ---- edge scatter: agg[col[e],:] += norm[e] * t[row[e],:]
// one wave per edge, 4 floats per lane (lane, lane+32, lane+64, lane+96)
__global__ __launch_bounds__(256) void scatter_kernel(
    const float* __restrict__ t, const long long* __restrict__ row,
    const long long* __restrict__ col, const float* __restrict__ norm,
    float* __restrict__ agg, int E) {
  int wv = threadIdx.x >> 5;
  int lane = threadIdx.x & 31;
  int e = blockIdx.x * 8 + wv;
  if (e >= E) return;
  int r = (int)row[e];
  int cnode = (int)col[e];
  float w = norm[e];
  const float* src = t + (size_t)r * HDIM + lane;
  float* dst = agg + (size_t)cnode * HDIM + lane;
#pragma unroll
  for (int i = 0; i < 4; ++i) atomicAdd(&dst[32 * i], w * src[32 * i]);
}

// ---- zero pooling accumulators
__global__ __launch_bounds__(256) void pool_zero_kernel(
    float* __restrict__ sums, float* __restrict__ cnt, int G) {
  int i = blockIdx.x * blockDim.x + threadIdx.x;
  if (i < G * HDIM) sums[i] = 0.f;
  if (i < G) cnt[i] = 0.f;
}

// ---- mean-pool accumulation with fused ReLU of the last layer
__global__ __launch_bounds__(128) void pool_kernel(
    const float* __restrict__ agg, const long long* __restrict__ batch,
    float* __restrict__ sums, float* __restrict__ cnt) {
  int n = blockIdx.x;
  int c = threadIdx.x;
  int g = (int)batch[n];
  float v = fmaxf(agg[(size_t)n * HDIM + c], 0.f);
  atomicAdd(&sums[g * HDIM + c], v);
  if (c == 0) atomicAdd(&cnt[g], 1.0f);
}

// ---- out[g] = (sums[g]/max(cnt,1)) . lin_w + lin_b
__global__ __launch_bounds__(128) void head_kernel(
    const float* __restrict__ sums, const float* __restrict__ cnt,
    const float* __restrict__ lin_w, const float* __restrict__ lin_b,
    float* __restrict__ out) {
  __shared__ float red[HDIM];
  int g = blockIdx.x;
  int c = threadIdx.x;
  float denom = fmaxf(cnt[g], 1.0f);
  red[c] = (sums[g * HDIM + c] / denom) * lin_w[c];
  __syncthreads();
#pragma unroll
  for (int s = 64; s > 0; s >>= 1) {
    if (c < s) red[c] += red[c + s];
    __syncthreads();
  }
  if (c == 0) out[g] = red[0] + lin_b[0];
}

extern "C" void kernel_launch(void* const* d_in, const int* in_sizes, int n_in,
                              void* d_out, int out_size, void* d_ws, size_t ws_size,
                              hipStream_t stream) {
  const float*     x     = (const float*)d_in[0];
  const float*     ew    = (const float*)d_in[1];
  const long long* ei    = (const long long*)d_in[2];   // int64 [2,E]
  const long long* batch = (const long long*)d_in[3];   // int64 [N]
  const float*     W1    = (const float*)d_in[4];
  const float*     b1    = (const float*)d_in[5];
  const float*     W2    = (const float*)d_in[6];
  const float*     b2    = (const float*)d_in[7];
  const float*     W3    = (const float*)d_in[8];
  const float*     b3    = (const float*)d_in[9];
  const float*     lin_w = (const float*)d_in[10];
  const float*     lin_b = (const float*)d_in[11];
  float*           out   = (float*)d_out;

  const int N = in_sizes[0] / HDIM;
  const int E = in_sizes[1];
  const int G = out_size;

  const long long* row = ei;       // source
  const long long* col = ei + E;   // target

  // ---- carve workspace (≈110 MB) -------------------------------------------
  char* p = (char*)d_ws;
  auto carve = [&](size_t bytes) -> char* {
    char* r = p;
    p += (bytes + 255) & ~(size_t)255;
    return r;
  };
  float*     t     = (float*)carve((size_t)N * HDIM * sizeof(float));
  float*     agg   = (float*)carve((size_t)N * HDIM * sizeof(float));
  float*     deg   = (float*)carve((size_t)N * sizeof(float));
  float*     dinv  = (float*)carve((size_t)N * sizeof(float));
  float*     selfn = (float*)carve((size_t)N * sizeof(float));
  float*     nrm   = (float*)carve((size_t)E * sizeof(float));
  _Float16*  Wt1   = (_Float16*)carve((size_t)HDIM * HDIM * sizeof(_Float16));
  _Float16*  Wt2   = (_Float16*)carve((size_t)HDIM * HDIM * sizeof(_Float16));
  _Float16*  Wt3   = (_Float16*)carve((size_t)HDIM * HDIM * sizeof(_Float16));
  float*     sums  = (float*)carve((size_t)G * HDIM * sizeof(float));
  float*     cnt   = (float*)carve((size_t)G * sizeof(float));
  (void)ws_size; (void)n_in;

  const int TB = 256;
  const int rowTiles = (N + 15) / 16;
  const int edgeBlocks = (E + 7) / 8;
  const int vecBlocks = (N * (HDIM / 4) + TB - 1) / TB;

  // ---- precompute -----------------------------------------------------------
  wt_convert_kernel<<<HDIM, HDIM, 0, stream>>>(W1, Wt1);
  wt_convert_kernel<<<HDIM, HDIM, 0, stream>>>(W2, Wt2);
  wt_convert_kernel<<<HDIM, HDIM, 0, stream>>>(W3, Wt3);
  deg_init_kernel<<<(N + TB - 1) / TB, TB, 0, stream>>>(deg, N);
  deg_accum_kernel<<<(E + TB - 1) / TB, TB, 0, stream>>>(col, ew, deg, E);
  dinv_kernel<<<(N + TB - 1) / TB, TB, 0, stream>>>(deg, dinv, selfn, N);
  norm_kernel<<<(E + TB - 1) / TB, TB, 0, stream>>>(row, col, ew, dinv, nrm, E);

  // ---- layer 1: input x (no input ReLU) ------------------------------------
  gcn_gemm_wmma_kernel<<<rowTiles, TB, 0, stream>>>(x, Wt1, t, 0, N);
  agg_init_kernel<<<vecBlocks, TB, 0, stream>>>(t, b1, selfn, agg, N);
  scatter_kernel<<<edgeBlocks, TB, 0, stream>>>(t, row, col, nrm, agg, E);

  // ---- layer 2: ReLU of layer-1 fused into A staging ------------------------
  gcn_gemm_wmma_kernel<<<rowTiles, TB, 0, stream>>>(agg, Wt2, t, 1, N);
  agg_init_kernel<<<vecBlocks, TB, 0, stream>>>(t, b2, selfn, agg, N);
  scatter_kernel<<<edgeBlocks, TB, 0, stream>>>(t, row, col, nrm, agg, E);

  // ---- layer 3 --------------------------------------------------------------
  gcn_gemm_wmma_kernel<<<rowTiles, TB, 0, stream>>>(agg, Wt3, t, 1, N);
  agg_init_kernel<<<vecBlocks, TB, 0, stream>>>(t, b3, selfn, agg, N);
  scatter_kernel<<<edgeBlocks, TB, 0, stream>>>(t, row, col, nrm, agg, E);

  // ---- pool (ReLU of layer-3 fused) + head ----------------------------------
  pool_zero_kernel<<<(G * HDIM + TB - 1) / TB, TB, 0, stream>>>(sums, cnt, G);
  pool_kernel<<<N, HDIM, 0, stream>>>(agg, batch, sums, cnt);
  head_kernel<<<G, HDIM, 0, stream>>>(sums, cnt, lin_w, lin_b, out);
}
